// VectorQuantizer_60507499266838
// MI455X (gfx1250) — matrix operand order repaired and checked by reading, not compiled, not run on hardware
//
#include <hip/hip_runtime.h>
#include <hip/hip_bf16.h>

typedef __attribute__((ext_vector_type(16))) __bf16 v16bf;
typedef __attribute__((ext_vector_type(4)))  __bf16 v4bf;
typedef __attribute__((ext_vector_type(8)))  float  v8f;

#define K_CODES 1024
#define DIM     64
#define N_ROWS  65536            // 64*32*32
#define N_ELEMS 4194304          // N_ROWS * DIM

// d_ws layout (bytes):
//   [0, 4096)     e2 : K f32  (|e_k|^2)
//   [4096, 4100)  acc: 1 f32  (loss accumulator)

__global__ __launch_bounds__(256)
void vq_prep(const float* __restrict__ cb, float* __restrict__ e2,
             float* __restrict__ acc) {
  int gid = blockIdx.x * blockDim.x + threadIdx.x;
  if (gid < K_CODES) {
    const float* r = cb + (size_t)gid * DIM;
    float s = 0.f;
    #pragma unroll
    for (int j = 0; j < DIM; ++j) s += r[j] * r[j];
    e2[gid] = s;
  }
  if (gid == 0) *acc = 0.f;
}

#define WMMA_BF16(A, B, C) \
  __builtin_amdgcn_wmma_f32_16x16x32_bf16(false, (A), false, (B), (short)0, (C), false, false)

__global__ __launch_bounds__(256)
void vq_main(const float* __restrict__ x, const float* __restrict__ cb,
             const float* __restrict__ e2g, float* __restrict__ out,
             float* __restrict__ acc) {
  // Whole bf16 hi/lo codebook lives in LDS: 2*128KB + 4KB <= 320KB/WGP.
  __shared__ __align__(32) __bf16 s_eh[K_CODES * DIM];
  __shared__ __align__(32) __bf16 s_el[K_CODES * DIM];
  __shared__ float s_e2[K_CODES];

  const int tid  = threadIdx.x;
  const int lane = threadIdx.x & 31;
  const int wave = threadIdx.x >> 5;
  const int h    = lane >> 4;        // lane half (0/1)
  const int m    = lane & 15;        // A: row in tile; B/C: column n
  const int rowBase = (blockIdx.x * 8 + wave) * 16;

  // ---- Stage codebook into LDS, converting f32 -> bf16 hi + bf16 residual.
  //      256 threads x 64 iters x 4 elems = 65536 elements, coalesced float4 reads.
  for (int c = 0; c < (K_CODES * DIM) / (256 * 4); ++c) {
    const int base = (c * 256 + tid) * 4;
    float4 v = *(const float4*)(cb + base);
    __bf16 h0 = (__bf16)v.x, h1 = (__bf16)v.y, h2 = (__bf16)v.z, h3 = (__bf16)v.w;
    v4bf hv = {h0, h1, h2, h3};
    v4bf lv = {(__bf16)(v.x - (float)h0), (__bf16)(v.y - (float)h1),
               (__bf16)(v.z - (float)h2), (__bf16)(v.w - (float)h3)};
    *(v4bf*)(s_eh + base) = hv;      // ds_store_b64
    *(v4bf*)(s_el + base) = lv;
  }
  for (int c = tid; c < K_CODES; c += 256) s_e2[c] = e2g[c];

  // ---- Build A tiles (hi/lo bf16 split), contiguous k layout (same as B):
  //      lane half h, pair p -> k = 2p + 16h  (chunk-local)
  v16bf ah0, ah1, al0, al1;
  const float* xrow = x + (size_t)(rowBase + m) * DIM;
  #pragma unroll
  for (int p = 0; p < 8; ++p) {
    const int k = 2 * p + 16 * h;
    float2 v0 = *(const float2*)(xrow + k);          // chunk 0: depth k, k+1
    __bf16 a0 = (__bf16)v0.x, a1 = (__bf16)v0.y;
    ah0[2 * p] = a0;                 ah0[2 * p + 1] = a1;
    al0[2 * p] = (__bf16)(v0.x - (float)a0);
    al0[2 * p + 1] = (__bf16)(v0.y - (float)a1);
    float2 v1 = *(const float2*)(xrow + 32 + k);     // chunk 1
    __bf16 b0 = (__bf16)v1.x, b1 = (__bf16)v1.y;
    ah1[2 * p] = b0;                 ah1[2 * p + 1] = b1;
    al1[2 * p] = (__bf16)(v1.x - (float)b0);
    al1[2 * p + 1] = (__bf16)(v1.y - (float)b1);
  }

  __syncthreads();

  float minv[8];
  int   mini[8];
  #pragma unroll
  for (int i = 0; i < 8; ++i) { minv[i] = 3.4e38f; mini[i] = 0; }

  // ---- Loop over 64 code tiles: score = |e|^2 - 2*(x . e).
  //      Two independent accumulator chains per tile; unroll 2 tiles so the
  //      next tile's ds_loads/WMMAs fill the WMMA->VALU hazard slots (v_nops).
  #pragma unroll 2
  for (int t = 0; t < K_CODES / 16; ++t) {
    const int    code = t * 16 + m;                  // this lane's column code
    const size_t boff = (size_t)code * DIM + 16 * h; // B layout: half h holds k=16h..16h+15
    v16bf bh0 = *(const v16bf*)(s_eh + boff);        // ds_load_b128 pairs
    v16bf bh1 = *(const v16bf*)(s_eh + boff + 32);
    v16bf bl0 = *(const v16bf*)(s_el + boff);
    v16bf bl1 = *(const v16bf*)(s_el + boff + 32);
    v8f c0 = {}, c1 = {};
    c0 = WMMA_BF16(ah0, bh0, c0);
    c1 = WMMA_BF16(ah1, bh1, c1);
    c0 = WMMA_BF16(al0, bh0, c0);
    c1 = WMMA_BF16(al1, bh1, c1);
    c0 = WMMA_BF16(ah0, bl0, c0);
    c1 = WMMA_BF16(ah1, bl1, c1);
    const float e2v = s_e2[code];
    #pragma unroll
    for (int i = 0; i < 8; ++i) {                    // C slot i -> row i + 8h
      float score = e2v - 2.0f * (c0[i] + c1[i]);
      if (score < minv[i]) { minv[i] = score; mini[i] = code; }
    }
  }

  // ---- Per-row argmin across the 16 lanes of each half (columns), tie -> lower idx
  #pragma unroll
  for (int i = 0; i < 8; ++i) {
    #pragma unroll
    for (int mask = 1; mask < 16; mask <<= 1) {
      float ov = __shfl_xor(minv[i], mask, 16);
      int   oi = __shfl_xor(mini[i], mask, 16);
      if (ov < minv[i] || (ov == minv[i] && oi < mini[i])) { minv[i] = ov; mini[i] = oi; }
    }
  }

  // ---- Gather codebook rows (f32), write quantized output, accumulate loss
  float lsum = 0.f;
  #pragma unroll
  for (int r = 0; r < 16; ++r) {
    const int idx = __shfl(mini[r & 7], (r >> 3) << 4, 32);   // row r lives in half r/8
    const float* q  = cb  + (size_t)idx * DIM;
    const float* xr = x   + (size_t)(rowBase + r) * DIM;
    float*       o  = out + (size_t)(rowBase + r) * DIM;
    float2 qv = *(const float2*)(q  + 2 * lane);
    float2 xv = *(const float2*)(xr + 2 * lane);
    float dx = qv.x - xv.x, dy = qv.y - xv.y;
    lsum += dx * dx + dy * dy;
    *(float2*)(o + 2 * lane) = qv;
  }
  #pragma unroll
  for (int mask = 1; mask < 32; mask <<= 1) lsum += __shfl_xor(lsum, mask, 32);
  if (lane == 0) atomicAdd(acc, lsum);
}

__global__ void vq_finalize(const float* __restrict__ acc, float* __restrict__ out) {
  // loss = q_latent + 0.25*e_latent = 1.25 * mean((q-x)^2)
  out[N_ELEMS] = 1.25f * (*acc) / (float)N_ELEMS;
}

extern "C" void kernel_launch(void* const* d_in, const int* in_sizes, int n_in,
                              void* d_out, int out_size, void* d_ws, size_t ws_size,
                              hipStream_t stream) {
  const float* x  = (const float*)d_in[0];   // [65536, 64]
  const float* cb = (const float*)d_in[1];   // [1024, 64]
  char* ws = (char*)d_ws;
  float* e2  = (float*)(ws);
  float* acc = (float*)(ws + 4096);
  float* out = (float*)d_out;

  vq_prep<<<4, 256, 0, stream>>>(cb, e2, acc);                       // |e|^2 + acc=0
  vq_main<<<N_ROWS / 128, 256, 0, stream>>>(x, cb, e2, out, acc);    // 512 blocks, 8 waves
  vq_finalize<<<1, 1, 0, stream>>>(acc, out);
}
